// SwigluMoEBlock_80083960201405
// MI455X (gfx1250) — matrix operand order repaired and checked by reading, not compiled, not run on hardware
//
#include <hip/hip_runtime.h>

// Problem constants (from reference setup_inputs)
#define ALPHA 1.702f
#define H_DIM 2048
#define I_DIM 2048
#define E_NUM 8
#define T_NUM 4096   // B*S = 4*1024

// GEMM tiling: block = 256 threads = 8 wave32; block tile 64x128, K-step 32.
// Wave w: m-tile = w>>1 (0..3), n-quad = (w&1)*4 -> 4 accumulators (A reuse x4).
#define BM 64
#define BN 128
#define BK 32
#define LDSK (BK + 8)   // padded LDS row (elements) to spread banks

typedef __attribute__((ext_vector_type(16))) __bf16 v16bf;
typedef __attribute__((ext_vector_type(8)))  float  v8f;
typedef __attribute__((ext_vector_type(2)))  float  v2f;
typedef __attribute__((ext_vector_type(2)))  __bf16 v2bf;
union Frag { v16bf v; uint4 q[2]; };

// Hardware packed f32->bf16 (RNE): lowers to v_cvt_pk_bf16_f32 on gfx1250
__device__ __forceinline__ unsigned bfpack2(float a, float b) {
  v2f f; f.x = a; f.y = b;
  v2bf h = __builtin_convertvector(f, v2bf);
  return __builtin_bit_cast(unsigned, h);
}
__device__ __forceinline__ unsigned short f2bf(float f) {
  return __builtin_bit_cast(unsigned short, (__bf16)f);
}

// ---------------- Gating: logits -> softmax -> top2 -> combine[T][8] -------
__global__ __launch_bounds__(256) void moe_gate(
    const float* __restrict__ x, const float* __restrict__ gw,
    float* __restrict__ combine)
{
  const int lane = threadIdx.x & 31;
  const int wv   = threadIdx.x >> 5;
  const int t    = blockIdx.x * 8 + wv;         // one wave32 per token
  const float* __restrict__ xr = x + (size_t)t * H_DIM;

  float acc[E_NUM];
  #pragma unroll
  for (int e = 0; e < E_NUM; ++e) acc[e] = 0.f;

  for (int h = lane; h < H_DIM; h += 32) {
    float xv = xr[h];
    #pragma unroll
    for (int e = 0; e < E_NUM; ++e)
      acc[e] = __fmaf_rn(xv, gw[(size_t)e * H_DIM + h], acc[e]);
  }
  #pragma unroll
  for (int e = 0; e < E_NUM; ++e) {
    float v = acc[e];
    #pragma unroll
    for (int off = 16; off > 0; off >>= 1) v += __shfl_xor(v, off, 32);
    acc[e] = v;
  }
  if (lane == 0) {
    float mx = acc[0];
    #pragma unroll
    for (int e = 1; e < E_NUM; ++e) mx = fmaxf(mx, acc[e]);
    float p[E_NUM];
    #pragma unroll
    for (int e = 0; e < E_NUM; ++e) p[e] = __expf(acc[e] - mx);
    // top-2 (first occurrence wins ties, matching lax.top_k)
    int i1 = 0;
    #pragma unroll
    for (int e = 1; e < E_NUM; ++e) if (p[e] > p[i1]) i1 = e;
    int i2 = (i1 == 0) ? 1 : 0;
    #pragma unroll
    for (int e = 0; e < E_NUM; ++e) if (e != i1 && p[e] > p[i2]) i2 = e;
    float s = p[i1] + p[i2];
    #pragma unroll
    for (int e = 0; e < E_NUM; ++e) {
      float w = (e == i1) ? p[i1] / s : ((e == i2) ? p[i2] / s : 0.f);
      combine[(size_t)t * E_NUM + e] = w;
    }
  }
}

// ---------------- fc1 + bias + SwiGLU -> a (bf16) --------------------------
__global__ __launch_bounds__(256) void moe_fc1_swiglu(
    const float* __restrict__ x, const float* __restrict__ w1,
    const float* __restrict__ b1, unsigned short* __restrict__ a_out, int e)
{
  __shared__ unsigned short As[BM][LDSK];
  __shared__ unsigned short Bs[BN][LDSK];
  __shared__ float Cs[BM][BN + 2];

  const int tid = threadIdx.x;
  const int bm0 = blockIdx.x * BM;
  const int bn0 = blockIdx.y * BN;               // index into 2*I features
  const float* __restrict__ wexp = w1 + (size_t)e * (2 * I_DIM) * H_DIM;

  const int lane  = tid & 31;
  const int wv    = tid >> 5;
  const int mt    = wv >> 1;                     // 0..3
  const int nq    = (wv & 1) * 4;                // 0 or 4
  const int l15   = lane & 15;
  const int khalf = lane >> 4;

  const int ca = tid & 7, ra = tid >> 3;         // f32 tile loader coords

  v8f acc[4] = {};

  // Software pipeline: registers hold tile k+1's global data while tile k computes.
  float4 aReg[2], bReg[4];
  #pragma unroll
  for (int i = 0; i < 2; ++i)
    aReg[i] = *(const float4*)(x + (size_t)(bm0 + ra + i * 32) * H_DIM + ca * 4);
  #pragma unroll
  for (int i = 0; i < 4; ++i)
    bReg[i] = *(const float4*)(wexp + (size_t)(bn0 + ra + i * 32) * H_DIM + ca * 4);

  for (int k0 = 0; k0 < H_DIM; k0 += BK) {
    // stage current tile (f32 regs -> bf16 LDS via v_cvt_pk_bf16_f32)
    #pragma unroll
    for (int i = 0; i < 2; ++i) {
      uint2 p; p.x = bfpack2(aReg[i].x, aReg[i].y); p.y = bfpack2(aReg[i].z, aReg[i].w);
      *(uint2*)&As[ra + i * 32][ca * 4] = p;
    }
    #pragma unroll
    for (int i = 0; i < 4; ++i) {
      uint2 p; p.x = bfpack2(bReg[i].x, bReg[i].y); p.y = bfpack2(bReg[i].z, bReg[i].w);
      *(uint2*)&Bs[ra + i * 32][ca * 4] = p;
    }
    __syncthreads();

    // issue next tile's global loads (overlap with fragment loads + WMMA)
    const int kn = k0 + BK;
    if (kn < H_DIM) {
      #pragma unroll
      for (int i = 0; i < 2; ++i)
        aReg[i] = *(const float4*)(x + (size_t)(bm0 + ra + i * 32) * H_DIM + kn + ca * 4);
      #pragma unroll
      for (int i = 0; i < 4; ++i)
        bReg[i] = *(const float4*)(wexp + (size_t)(bn0 + ra + i * 32) * H_DIM + kn + ca * 4);
    }

    // A fragment: lane holds row (mt*16+l15); K halves 0-7/16-23 or 8-15/24-31
    Frag fa;
    fa.q[0] = *(const uint4*)&As[mt * 16 + l15][khalf * 8];
    fa.q[1] = *(const uint4*)&As[mt * 16 + l15][16 + khalf * 8];
    // prefetch all B fragments, then issue WMMAs back-to-back
    Frag fb[4];
    #pragma unroll
    for (int j = 0; j < 4; ++j) {
      fb[j].q[0] = *(const uint4*)&Bs[(nq + j) * 16 + l15][khalf * 8];
      fb[j].q[1] = *(const uint4*)&Bs[(nq + j) * 16 + l15][16 + khalf * 8];
    }
    #pragma unroll
    for (int j = 0; j < 4; ++j)
      acc[j] = __builtin_amdgcn_wmma_f32_16x16x32_bf16(
          false, fa.v, false, fb[j].v, (short)0, acc[j], false, false);
    __syncthreads();
  }

  // Spill f32 tile to LDS so SwiGLU pairing (2i,2i+1) is local
  #pragma unroll
  for (int j = 0; j < 4; ++j)
    #pragma unroll
    for (int v = 0; v < 8; ++v)
      Cs[mt * 16 + khalf * 8 + v][(nq + j) * 16 + l15] = acc[j][v];
  __syncthreads();

  const float* __restrict__ brow = b1 + (size_t)e * (2 * I_DIM) + bn0;
  for (int idx = tid; idx < BM * (BN / 2); idx += 256) {
    int r = idx >> 6;
    int c = idx & 63;
    float g = Cs[r][2 * c]     + brow[2 * c];
    float l = Cs[r][2 * c + 1] + brow[2 * c + 1];
    float av = g / (1.f + __expf(-ALPHA * g)) * (l + 1.f);
    a_out[(size_t)(bm0 + r) * I_DIM + (bn0 >> 1) + c] = f2bf(av);
  }
}

// ---------------- fc2 + bias + combine-weighted accumulate into out --------
__global__ __launch_bounds__(256) void moe_fc2_combine(
    const unsigned short* __restrict__ a_in, const float* __restrict__ w2,
    const float* __restrict__ b2, const float* __restrict__ combine,
    float* __restrict__ out, int e)
{
  __shared__ unsigned short As[BM][LDSK];
  __shared__ unsigned short Bs[BN][LDSK];

  const int tid = threadIdx.x;
  const int bm0 = blockIdx.x * BM;
  const int bn0 = blockIdx.y * BN;               // h (output) dimension
  const float* __restrict__ wexp = w2 + (size_t)e * H_DIM * I_DIM;

  const int lane  = tid & 31;
  const int wv    = tid >> 5;
  const int mt    = wv >> 1;
  const int nq    = (wv & 1) * 4;
  const int l15   = lane & 15;
  const int khalf = lane >> 4;

  const int ca = tid & 7, ra = tid >> 3;         // f32 B loader
  const int cb = tid & 3, rb = tid >> 2;         // bf16 A loader (16B chunks)

  v8f acc[4] = {};

  // Software pipeline registers
  uint4  aReg;
  float4 bReg[4];
  aReg = *(const uint4*)(a_in + (size_t)(bm0 + rb) * I_DIM + cb * 8);
  #pragma unroll
  for (int i = 0; i < 4; ++i)
    bReg[i] = *(const float4*)(wexp + (size_t)(bn0 + ra + i * 32) * I_DIM + ca * 4);

  for (int k0 = 0; k0 < I_DIM; k0 += BK) {
    *(uint4*)&As[rb][cb * 8] = aReg;
    #pragma unroll
    for (int i = 0; i < 4; ++i) {
      uint2 p; p.x = bfpack2(bReg[i].x, bReg[i].y); p.y = bfpack2(bReg[i].z, bReg[i].w);
      *(uint2*)&Bs[ra + i * 32][ca * 4] = p;
    }
    __syncthreads();

    const int kn = k0 + BK;
    if (kn < I_DIM) {
      aReg = *(const uint4*)(a_in + (size_t)(bm0 + rb) * I_DIM + kn + cb * 8);
      #pragma unroll
      for (int i = 0; i < 4; ++i)
        bReg[i] = *(const float4*)(wexp + (size_t)(bn0 + ra + i * 32) * I_DIM + kn + ca * 4);
    }

    Frag fa;
    fa.q[0] = *(const uint4*)&As[mt * 16 + l15][khalf * 8];
    fa.q[1] = *(const uint4*)&As[mt * 16 + l15][16 + khalf * 8];
    Frag fb[4];
    #pragma unroll
    for (int j = 0; j < 4; ++j) {
      fb[j].q[0] = *(const uint4*)&Bs[(nq + j) * 16 + l15][khalf * 8];
      fb[j].q[1] = *(const uint4*)&Bs[(nq + j) * 16 + l15][16 + khalf * 8];
    }
    #pragma unroll
    for (int j = 0; j < 4; ++j)
      acc[j] = __builtin_amdgcn_wmma_f32_16x16x32_bf16(
          false, fa.v, false, fb[j].v, (short)0, acc[j], false, false);
    __syncthreads();
  }

  // Epilogue: out[t,h] (+)= combine[t,e] * (acc + fc2_b[e,h]).
  // e==0 overwrites (d_out is poisoned), e>0 accumulates -> deterministic.
  float wrow[8];
  #pragma unroll
  for (int v = 0; v < 8; ++v)
    wrow[v] = combine[(size_t)(bm0 + mt * 16 + khalf * 8 + v) * E_NUM + e];

  #pragma unroll
  for (int j = 0; j < 4; ++j) {
    int ncol = bn0 + (nq + j) * 16 + l15;
    float bias = b2[(size_t)e * H_DIM + ncol];
    #pragma unroll
    for (int v = 0; v < 8; ++v) {
      size_t oidx = (size_t)(bm0 + mt * 16 + khalf * 8 + v) * H_DIM + ncol;
      float prev = (e == 0) ? 0.f : out[oidx];
      out[oidx] = prev + wrow[v] * (acc[j][v] + bias);
    }
  }
}

// ---------------------------------------------------------------------------
extern "C" void kernel_launch(void* const* d_in, const int* in_sizes, int n_in,
                              void* d_out, int out_size, void* d_ws, size_t ws_size,
                              hipStream_t stream) {
  const float* x  = (const float*)d_in[0];
  const float* gw = (const float*)d_in[1];
  const float* w1 = (const float*)d_in[2];
  const float* b1 = (const float*)d_in[3];
  const float* w2 = (const float*)d_in[4];
  const float* b2 = (const float*)d_in[5];
  float* out = (float*)d_out;

  // ws: combine[T][E] f32 (128 KB) | a_buf[T][I] bf16 (16 MB)
  float* combine = (float*)d_ws;
  size_t comb_bytes = (((size_t)T_NUM * E_NUM * sizeof(float)) + 255) & ~(size_t)255;
  unsigned short* a_buf = (unsigned short*)((char*)d_ws + comb_bytes);

  moe_gate<<<T_NUM / 8, 256, 0, stream>>>(x, gw, combine);
  for (int e = 0; e < E_NUM; ++e) {
    moe_fc1_swiglu<<<dim3(T_NUM / BM, (2 * I_DIM) / BN), 256, 0, stream>>>(
        x, w1, b1, a_buf, e);
    moe_fc2_combine<<<dim3(T_NUM / BM, H_DIM / BN), 256, 0, stream>>>(
        a_buf, w2, b2, combine, out, e);
  }
}
